// CRF_29308856828584
// MI455X (gfx1250) — matrix operand order repaired and checked by reading, not compiled, not run on hardware
//
#include <hip/hip_runtime.h>

typedef __attribute__((ext_vector_type(16))) _Float16 v16h;
typedef __attribute__((ext_vector_type(8)))  float    v8f;

union V16H { v16h h; unsigned u[8]; };

#define LOG2E 1.4426950408889634f
#define LN2   0.6931471805599453f

__device__ __forceinline__ unsigned pk(float a, float b) {
  auto r = __builtin_amdgcn_cvt_pkrtz(a, b);  // v_cvt_pk_f16_f32: a->[15:0], b->[31:16]
  return __builtin_bit_cast(unsigned, r);
}

// One wave32 owns a 16-row batch tile. alpha kept in D-layout registers:
//   al[tt][r]  <->  state j = 16*tt + 8*g + r, batch column b = lane&15.
__global__ __launch_bounds__(32, 1) void crf_fwd_kernel(
    const float* __restrict__ unary, const int* __restrict__ lengths,
    const float* __restrict__ trans, float* __restrict__ out, int T)
{
  const int lane = threadIdx.x & 31;
  const int b    = lane & 15;
  const int g    = lane >> 4;
  const int b0   = blockIdx.x * 16;

  // 8-deep ring of u_t tiles: slot s holds 16 rows x 64 floats (4 KB), [brow][j].
  __shared__ float ldsbuf[8 * 1024];
  const unsigned ldsBase = (unsigned)(uintptr_t)&ldsbuf[0];

  // ---- constant A operands: Etr[tt][k] = 16x32 tile of E^T, E = exp(trans).
  // A operand, this code's contiguous K order: row M = lane&15, VGPR v holds K pair
  // {2v, 2v+1} + 16*g.  E^T[j,i] = exp(trans[i*64+j]).
  V16H Etr[4][2];
  #pragma unroll
  for (int tt = 0; tt < 4; ++tt) {
    const int j = 16*tt + b;
    #pragma unroll
    for (int k = 0; k < 2; ++k)
      #pragma unroll
      for (int v = 0; v < 8; ++v) {
        const int i0 = 2*v + 16*g + 32*k;
        Etr[tt][k].u[v] = pk(__builtin_amdgcn_exp2f(trans[(i0  )*64 + j] * LOG2E),
                             __builtin_amdgcn_exp2f(trans[(i0+1)*64 + j] * LOG2E));
      }
  }

  const int len = lengths[b0 + b];

  // Async staging: chunk kc copies rows {2kc, 2kc+1}: lanes 0-15 -> row 2kc,
  // lanes 16-31 -> row 2kc+1, 16 B per lane => one fully-coalesced 512 B op.
  const unsigned rowBase   = (unsigned)(b0 + g) * (unsigned)(T * 256) + (unsigned)(b * 16);
  const unsigned rowPair   = (unsigned)(2 * T * 256);
  const unsigned ldsLaneOf = (unsigned)(b * 16 + g * 256);

  auto issue_slot = [&](int tsrc, int slot) {
    const unsigned lo = ldsBase + (unsigned)slot * 4096u + ldsLaneOf;
    const unsigned to = (unsigned)tsrc * 256u;
    #pragma unroll
    for (int kc = 0; kc < 8; ++kc) {
      unsigned ldsoff = lo + (unsigned)kc * 512u;
      unsigned voff   = rowBase + (unsigned)kc * rowPair + to;
      asm volatile("global_load_async_to_lds_b128 %0, %1, %2"
                   :: "v"(ldsoff), "v"(voff), "s"(unary) : "memory");
    }
  };

  // ---- alpha init from t = 0 (direct register loads, one time)
  const float4* __restrict__ up =
      reinterpret_cast<const float4*>(unary) + (size_t)(b0 + b) * T * 16;
  float al[4][8];
  #pragma unroll
  for (int tt = 0; tt < 4; ++tt) {
    float4 lo = up[4*tt + 2*g];
    float4 hi = up[4*tt + 2*g + 1];
    al[tt][0]=lo.x; al[tt][1]=lo.y; al[tt][2]=lo.z; al[tt][3]=lo.w;
    al[tt][4]=hi.x; al[tt][5]=hi.y; al[tt][6]=hi.z; al[tt][7]=hi.w;
  }

  // ---- prologue: fill the ring with u_1 .. u_8
  #pragma unroll
  for (int s = 0; s < 8; ++s) {
    int ts = 1 + s; if (ts > T-1) ts = T-1;
    issue_slot(ts, (1 + s) & 7);
  }

  for (int t = 1; t < T; ++t) {
    // oldest slot (for this t) has landed once <= 56 async ops outstanding
    asm volatile("s_wait_asynccnt 0x38" ::: "memory");

    // ---- m_b = max_j alpha[j,b]  (own 32 states, then partner half-wave)
    float m = al[0][0];
    #pragma unroll
    for (int tt = 0; tt < 4; ++tt)
      #pragma unroll
      for (int r = 0; r < 8; ++r) m = fmaxf(m, al[tt][r]);
    m = fmaxf(m, __shfl_xor(m, 16, 32));
    const float nml = -m * LOG2E;   // exp(al-m) = exp2(fma(al, log2e, nml))

    // ---- p = exp(alpha - m) in (0,1], packed f16 pairs (consecutive rows)
    unsigned ph[4][4];
    #pragma unroll
    for (int tt = 0; tt < 4; ++tt)
      #pragma unroll
      for (int q = 0; q < 4; ++q)
        ph[tt][q] = pk(__builtin_amdgcn_exp2f(fmaf(al[tt][2*q  ], LOG2E, nml)),
                       __builtin_amdgcn_exp2f(fmaf(al[tt][2*q+1], LOG2E, nml)));

    // ---- B operands (P^T): every needed value is lane-local or from lane^16.
    unsigned phx[4][4];
    #pragma unroll
    for (int tt = 0; tt < 4; ++tt)
      #pragma unroll
      for (int q = 0; q < 4; ++q) phx[tt][q] = __shfl_xor(ph[tt][q], 16, 32);

    V16H B0, B1;
    #pragma unroll
    for (int q = 0; q < 4; ++q) {
      B0.u[q]   = g ? phx[1][q] : ph [0][q];
      B0.u[q+4] = g ? ph [1][q] : phx[0][q];
      B1.u[q]   = g ? phx[3][q] : ph [2][q];
      B1.u[q+4] = g ? ph [3][q] : phx[2][q];
    }

    // ---- Q^T = E^T * P^T : 4 output tiles x 2 K-steps = 8 WMMAs/step
    v8f acc[4];
    #pragma unroll
    for (int tt = 0; tt < 4; ++tt) {
      v8f c = {0.f,0.f,0.f,0.f,0.f,0.f,0.f,0.f};
      c = __builtin_amdgcn_wmma_f32_16x16x32_f16(false, Etr[tt][0].h, false, B0.h,
                                                 (short)0, c, false, false);
      acc[tt] = __builtin_amdgcn_wmma_f32_16x16x32_f16(false, Etr[tt][1].h, false, B1.h,
                                                       (short)0, c, false, false);
    }

    // ---- u_t from the LDS ring (b128 DS reads), then alpha update
    const bool keep = (t < len);
    const float* sp = &ldsbuf[(size_t)(t & 7) * 1024 + b * 64 + 8 * g];
    #pragma unroll
    for (int tt = 0; tt < 4; ++tt) {
      const float4 u0 = ((const float4*)(sp + 16*tt))[0];
      const float4 u1 = ((const float4*)(sp + 16*tt))[1];
      #pragma unroll
      for (int r = 0; r < 8; ++r) {
        const float uu = (r < 4) ? (&u0.x)[r] : (&u1.x)[r-4];
        // q in [~0.6, ~106]: raw v_log_f32 is safe; an = m + uu + ln2*log2(q)
        const float an = fmaf(LN2, __builtin_amdgcn_logf(acc[tt][r]), m + uu);
        al[tt][r] = keep ? an : al[tt][r];        // jnp.where(t < len, ...)
      }
    }

    // all DS reads of this slot done before the async overwrite lands
    asm volatile("s_wait_dscnt 0x0" ::: "memory");
    int tnext = t + 8; if (tnext > T-1) tnext = T-1;
    issue_slot(tnext, (t + 8) & 7);
  }

  // ---- out[b] = logsumexp_j alpha[b,j]
  float m = al[0][0];
  #pragma unroll
  for (int tt = 0; tt < 4; ++tt)
    #pragma unroll
    for (int r = 0; r < 8; ++r) m = fmaxf(m, al[tt][r]);
  m = fmaxf(m, __shfl_xor(m, 16, 32));
  const float nml = -m * LOG2E;
  float s = 0.f;
  #pragma unroll
  for (int tt = 0; tt < 4; ++tt)
    #pragma unroll
    for (int r = 0; r < 8; ++r)
      s += __builtin_amdgcn_exp2f(fmaf(al[tt][r], LOG2E, nml));
  s += __shfl_xor(s, 16, 32);
  if (g == 0) out[b0 + b] = fmaf(LN2, __builtin_amdgcn_logf(s), m);
}

extern "C" void kernel_launch(void* const* d_in, const int* in_sizes, int n_in,
                              void* d_out, int out_size, void* d_ws, size_t ws_size,
                              hipStream_t stream) {
  const float* unary   = (const float*)d_in[0];   // (B, T, 64) f32
  const int*   lengths = (const int*)  d_in[1];   // (B,) i32
  const float* trans   = (const float*)d_in[2];   // (64, 64) f32
  float*       out     = (float*)d_out;           // (B,) f32

  const int B = in_sizes[1];
  const int T = in_sizes[0] / (B * 64);

  dim3 grid(B / 16), block(32);                   // one wave32 per 16-batch tile
  crf_fwd_kernel<<<grid, block, 0, stream>>>(unary, lengths, trans, out, T);
}
